// MNIST_Graph_65249143160992
// MI455X (gfx1250) — matrix operand order repaired and checked by reading, not compiled, not run on hardware
//
#include <hip/hip_runtime.h>
#include <hip/hip_fp16.h>

typedef __attribute__((ext_vector_type(16))) _Float16 v16h;
typedef __attribute__((ext_vector_type(8)))  float    v8f;

#define LDW 20      // LDS tile row stride (floats)
#define WPB 8       // waves per 256-thread block (wave32)

// ---------------------------------------------------------------------------
// WMMA helper: D = A(16x32 f16) x B(32x16 f16) + C(16x16 f32)
// ---------------------------------------------------------------------------
static __device__ __forceinline__ v8f wmma16(v16h a, v16h b, v8f c) {
  return __builtin_amdgcn_wmma_f32_16x16x32_f16(false, a, false, b, (short)0, c,
                                                false, false);
}

// Branchless SELU: selu(x) = l*max(x,0) + l*a*(exp(min(x,0)) - 1)
// Exact: for x>0 the exp term vanishes; for x<=0 the max term vanishes.
// Lowers to straight-line VALU (no EXEC-mask branching).
static __device__ __forceinline__ float selu1(float x) {
  const float l  = 1.0507009873554805f;
  const float la = 1.7580993408473766f;  // l * 1.6732632423543772
  return fmaf(l, fmaxf(x, 0.f), la * (__expf(fminf(x, 0.f)) - 1.f));
}
static __device__ __forceinline__ v8f selu8(v8f x) {
  v8f r;
#pragma unroll
  for (int i = 0; i < 8; ++i) r[i] = selu1(x[i]);
  return r;
}

// Build B operand (32x16 f16, KxN) from row-major f32 weights W[K][N].
// ISA B layout: lane n (and n+16) hold column n; lanes<16: K=0..7 (v0..3), 16..23 (v4..7);
// lanes>=16: K=8..15 (v0..3), 24..31 (v4..7).
static __device__ __forceinline__ v16h make_B(const float* W, int K, int N, int lane) {
  int col = lane & 15;
  int kb  = (lane < 16) ? 0 : 8;
  v16h b = {};
#pragma unroll
  for (int j = 0; j < 8; ++j) {
    int k0 = (j < 4) ? (kb + 2 * j) : (kb + 16 + 2 * (j - 4));
    float w0 = (k0     < K && col < N) ? W[k0 * N + col]       : 0.f;
    float w1 = (k0 + 1 < K && col < N) ? W[(k0 + 1) * N + col] : 0.f;
    b[2 * j]     = (_Float16)w0;
    b[2 * j + 1] = (_Float16)w1;
  }
  return b;
}

// Bias broadcast in C layout (every row gets bias[col]).
static __device__ __forceinline__ v8f make_biasC(const float* bias, int N, int lane) {
  int col = lane & 15;
  float v = (col < N) ? bias[col] : 0.f;
  v8f c;
#pragma unroll
  for (int i = 0; i < 8; ++i) c[i] = v;
  return c;
}

// Store C tile (16x16 f32) into LDS at column offset co.
// C layout: lane l holds col N=l%16, rows M = d + 8*(l>=16), d=0..7.
static __device__ __forceinline__ void storeC(float* tile, v8f c, int co, int lane) {
  int col = lane & 15;
  int mb  = (lane < 16) ? 0 : 8;
#pragma unroll
  for (int d = 0; d < 8; ++d) tile[(mb + d) * LDW + co + col] = c[d];
}

// Load A operand (16x32 f16) from a row-major LDS tile (stride LDW), K valid cols.
// ISA A layout: lane m (and m+16) hold row m; lanes<16: K=0..7 (v0..3), 16..23 (v4..7);
// lanes>=16: K=8..15 (v0..3), 24..31 (v4..7).
static __device__ __forceinline__ v16h loadA(const float* tile, int K, int lane) {
  int row = lane & 15;
  int kb  = (lane < 16) ? 0 : 8;
  v16h a = {};
#pragma unroll
  for (int j = 0; j < 8; ++j) {
    int k = (j < 4) ? (kb + 2 * j) : (kb + 16 + 2 * (j - 4));
    float x0 = (k     < K) ? tile[row * LDW + k]     : 0.f;
    float x1 = (k + 1 < K) ? tile[row * LDW + k + 1] : 0.f;
    a[2 * j]     = (_Float16)x0;
    a[2 * j + 1] = (_Float16)x1;
  }
  return a;
}

// ---------------------------------------------------------------------------
__global__ void __launch_bounds__(256) zero_kernel(float* p, int n) {
  int i = blockIdx.x * blockDim.x + threadIdx.x;
  if (i < n) p[i] = 0.f;
}

// ---------------------------------------------------------------------------
// Kernel 1: per-edge MLP chain + segment-sum into edge_accum[receiver]
// ---------------------------------------------------------------------------
__global__ void __launch_bounds__(256) edge_kernel(
    const float* __restrict__ nodes, const float* __restrict__ edges,
    const int* __restrict__ senders, const int* __restrict__ receivers,
    const float* __restrict__ pn_w1, const float* __restrict__ pn_b1,
    const float* __restrict__ pn_w2, const float* __restrict__ pn_b2,
    const float* __restrict__ ue_w1, const float* __restrict__ ue_b1,
    const float* __restrict__ ue_w2, const float* __restrict__ ue_b2,
    float* __restrict__ edge_accum, int E) {
  __shared__ float lds[WPB][2][16 * LDW];
  const int lane = threadIdx.x & 31;
  const int wv   = threadIdx.x >> 5;
  float* tileH = &lds[wv][0][0];
  float* tileT = &lds[wv][1][0];
  const int col = lane & 15;

  v16h B1  = make_B(pn_w1, 2, 10, lane);
  v16h B2  = make_B(pn_w2, 10, 10, lane);
  v16h Bu1 = make_B(ue_w1, 11, 5, lane);
  v16h Bu2 = make_B(ue_w2, 5, 1, lane);
  v8f  C1  = make_biasC(pn_b1, 10, lane);
  v8f  C2  = make_biasC(pn_b2, 10, lane);
  v8f  Cu1 = make_biasC(ue_b1, 5, lane);
  v8f  Cu2 = make_biasC(ue_b2, 1, lane);

  int nTiles  = (E + 15) >> 4;
  int wgid    = blockIdx.x * WPB + wv;
  int wstride = gridDim.x * WPB;
  for (int t = wgid; t < nTiles; t += wstride) {
    int eidx = t * 16 + col;
    bool rowValid = (lane < 16) && (eidx < E);
    int r = 0;
    float xr = 0.f, xs = 0.f, ev = 0.f;
    if (rowValid) {
      r = receivers[eidx];
      int s = senders[eidx];
      xr = nodes[r];
      xs = nodes[s];
      ev = edges[eidx];
    }
    // permutate_nodes: [nodes[recv] || nodes[send]] -> 10 (SELU) -> 10
    v16h A = {};
    if (lane < 16) { A[0] = (_Float16)xr; A[1] = (_Float16)xs; }
    v8f h = selu8(wmma16(A, B1, C1));
    storeC(tileT, h, 0, lane);
    v8f hn = wmma16(loadA(tileT, 10, lane), B2, C2);  // h1_nodes, no out activation
    storeC(tileH, hn, 1, lane);                        // cols 1..10
    if (lane < 16) tileH[lane * LDW + 0] = ev;         // col 0 = h1_edges

    // 3x update_edges: concat(h_e, h1_nodes) 11 -> 5 (SELU) -> 1
#pragma unroll
    for (int it = 0; it < 3; ++it) {
      v8f u = selu8(wmma16(loadA(tileH, 11, lane), Bu1, Cu1));
      storeC(tileT, u, 0, lane);
      v8f w = wmma16(loadA(tileT, 5, lane), Bu2, Cu2);
      if (col == 0) {
        int mb = (lane < 16) ? 0 : 8;
#pragma unroll
        for (int d = 0; d < 8; ++d) tileH[(mb + d) * LDW + 0] = w[d];
      }
    }
    if (rowValid) atomicAdd(&edge_accum[r], tileH[lane * LDW + 0]);
  }
}

// ---------------------------------------------------------------------------
// Kernel 2: per-node MLP chain + segment-sum of concat(h2_nodes, h2_edges)
// ---------------------------------------------------------------------------
__global__ void __launch_bounds__(256) node_kernel(
    const float* __restrict__ nodes, const float* __restrict__ edge_accum,
    const int* __restrict__ graph_ids,
    const float* __restrict__ pe_w1, const float* __restrict__ pe_b1,
    const float* __restrict__ pe_w2, const float* __restrict__ pe_b2,
    const float* __restrict__ un_w1, const float* __restrict__ un_b1,
    const float* __restrict__ un_w2, const float* __restrict__ un_b2,
    float* __restrict__ graph_accum, int N) {
  __shared__ float lds[WPB][2][16 * LDW];
  const int lane = threadIdx.x & 31;
  const int wv   = threadIdx.x >> 5;
  float* tileH = &lds[wv][0][0];
  float* tileT = &lds[wv][1][0];
  const int col = lane & 15;

  v16h B1  = make_B(pe_w1, 1, 5, lane);
  v16h B2  = make_B(pe_w2, 5, 10, lane);
  v16h Bu1 = make_B(un_w1, 11, 5, lane);
  v16h Bu2 = make_B(un_w2, 5, 1, lane);
  v8f  C1  = make_biasC(pe_b1, 5, lane);
  v8f  C2  = make_biasC(pe_b2, 10, lane);
  v8f  Cu1 = make_biasC(un_b1, 5, lane);
  v8f  Cu2 = make_biasC(un_b2, 1, lane);

  int nTiles  = (N + 15) >> 4;
  int wgid    = blockIdx.x * WPB + wv;
  int wstride = gridDim.x * WPB;
  for (int t = wgid; t < nTiles; t += wstride) {
    int nidx = t * 16 + col;
    bool rowValid = (lane < 16) && (nidx < N);
    float ea = 0.f, xv = 0.f;
    int gid = 0;
    if (rowValid) {
      ea = edge_accum[nidx];
      xv = nodes[nidx];
      gid = graph_ids[nidx];
    }
    // permutate_edges: 1 -> 5 (SELU) -> 10
    v16h A = {};
    if (lane < 16) A[0] = (_Float16)ea;
    v8f h = selu8(wmma16(A, B1, C1));
    storeC(tileT, h, 0, lane);
    v8f he = wmma16(loadA(tileT, 5, lane), B2, C2);  // h2_edges
    storeC(tileH, he, 1, lane);                       // cols 1..10
    if (lane < 16) tileH[lane * LDW + 0] = xv;        // col 0 = h2_nodes

    // 3x update_nodes: concat(h_n, h2_edges) 11 -> 5 (SELU) -> 1
#pragma unroll
    for (int it = 0; it < 3; ++it) {
      v8f u = selu8(wmma16(loadA(tileH, 11, lane), Bu1, Cu1));
      storeC(tileT, u, 0, lane);
      v8f w = wmma16(loadA(tileT, 5, lane), Bu2, Cu2);
      if (col == 0) {
        int mb = (lane < 16) ? 0 : 8;
#pragma unroll
        for (int d = 0; d < 8; ++d) tileH[(mb + d) * LDW + 0] = w[d];
      }
    }
    if (rowValid) {
#pragma unroll
      for (int j = 0; j < 11; ++j)
        atomicAdd(&graph_accum[gid * 11 + j], tileH[lane * LDW + j]);
    }
  }
}

// ---------------------------------------------------------------------------
// Kernel 3: per-graph head: 11 -> 10 (SELU) -> 10 -> softmax
// ---------------------------------------------------------------------------
__global__ void __launch_bounds__(256) graph_kernel(
    const float* __restrict__ graph_accum,
    const float* __restrict__ pr_w1, const float* __restrict__ pr_b1,
    const float* __restrict__ pr_w2, const float* __restrict__ pr_b2,
    float* __restrict__ out, int G) {
  __shared__ float lds[WPB][16 * LDW];
  const int lane = threadIdx.x & 31;
  const int wv   = threadIdx.x >> 5;
  float* tileT = &lds[wv][0];

  v16h B1 = make_B(pr_w1, 11, 10, lane);
  v16h B2 = make_B(pr_w2, 10, 10, lane);
  v8f  C1 = make_biasC(pr_b1, 10, lane);
  v8f  C2 = make_biasC(pr_b2, 10, lane);

  int nTiles  = (G + 15) >> 4;
  int wgid    = blockIdx.x * WPB + wv;
  int wstride = gridDim.x * WPB;
  int row = lane & 15;
  for (int t = wgid; t < nTiles; t += wstride) {
    int g = t * 16 + row;
    v16h A = {};
    if (g < G) {
      int kb = (lane < 16) ? 0 : 8;
#pragma unroll
      for (int j = 0; j < 4; ++j) {  // K>=16 slots are zero (K=11)
        int k = kb + 2 * j;
        float x0 = (k     < 11) ? graph_accum[g * 11 + k]     : 0.f;
        float x1 = (k + 1 < 11) ? graph_accum[g * 11 + k + 1] : 0.f;
        A[2 * j]     = (_Float16)x0;
        A[2 * j + 1] = (_Float16)x1;
      }
    }
    v8f h = selu8(wmma16(A, B1, C1));
    storeC(tileT, h, 0, lane);
    v8f o = wmma16(loadA(tileT, 10, lane), B2, C2);
    storeC(tileT, o, 0, lane);
    if ((lane < 16) && (g < G)) {
      float v[10], m = -3.4e38f;
#pragma unroll
      for (int j = 0; j < 10; ++j) { v[j] = tileT[lane * LDW + j]; m = fmaxf(m, v[j]); }
      float s = 0.f;
#pragma unroll
      for (int j = 0; j < 10; ++j) { v[j] = __expf(v[j] - m); s += v[j]; }
      float inv = 1.f / s;
#pragma unroll
      for (int j = 0; j < 10; ++j) out[g * 10 + j] = v[j] * inv;
    }
  }
}

// ---------------------------------------------------------------------------
extern "C" void kernel_launch(void* const* d_in, const int* in_sizes, int n_in,
                              void* d_out, int out_size, void* d_ws, size_t ws_size,
                              hipStream_t stream) {
  const float* nodes     = (const float*)d_in[0];
  const float* edges     = (const float*)d_in[1];
  const int*   senders   = (const int*)d_in[2];
  const int*   receivers = (const int*)d_in[3];
  const int*   graph_ids = (const int*)d_in[4];
  // d_in[5] = num_graphs scalar (G derived from out_size instead)
  const float* pn_w1 = (const float*)d_in[6],  *pn_b1 = (const float*)d_in[7];
  const float* pn_w2 = (const float*)d_in[8],  *pn_b2 = (const float*)d_in[9];
  const float* ue_w1 = (const float*)d_in[10], *ue_b1 = (const float*)d_in[11];
  const float* ue_w2 = (const float*)d_in[12], *ue_b2 = (const float*)d_in[13];
  const float* pe_w1 = (const float*)d_in[14], *pe_b1 = (const float*)d_in[15];
  const float* pe_w2 = (const float*)d_in[16], *pe_b2 = (const float*)d_in[17];
  const float* un_w1 = (const float*)d_in[18], *un_b1 = (const float*)d_in[19];
  const float* un_w2 = (const float*)d_in[20], *un_b2 = (const float*)d_in[21];
  const float* pr_w1 = (const float*)d_in[22], *pr_b1 = (const float*)d_in[23];
  const float* pr_w2 = (const float*)d_in[24], *pr_b2 = (const float*)d_in[25];

  const int N = in_sizes[0];
  const int E = in_sizes[1];
  const int G = out_size / 10;
  float* out = (float*)d_out;

  float* edge_accum  = (float*)d_ws;       // N floats
  float* graph_accum = edge_accum + N;     // G*11 floats

  int zn = N + G * 11;
  zero_kernel<<<(zn + 255) / 256, 256, 0, stream>>>(edge_accum, zn);

  int etiles = (E + 15) / 16;
  edge_kernel<<<(etiles + WPB - 1) / WPB, 256, 0, stream>>>(
      nodes, edges, senders, receivers,
      pn_w1, pn_b1, pn_w2, pn_b2, ue_w1, ue_b1, ue_w2, ue_b2,
      edge_accum, E);

  int ntiles = (N + 15) / 16;
  node_kernel<<<(ntiles + WPB - 1) / WPB, 256, 0, stream>>>(
      nodes, edge_accum, graph_ids,
      pe_w1, pe_b1, pe_w2, pe_b2, un_w1, un_b1, un_w2, un_b2,
      graph_accum, N);

  int gtiles = (G + 15) / 16;
  graph_kernel<<<(gtiles + WPB - 1) / WPB, 256, 0, stream>>>(
      graph_accum, pr_w1, pr_b1, pr_w2, pr_b2, out, G);
}
